// MultiHeadAttention_44272522887605
// MI455X (gfx1250) — matrix operand order repaired and checked by reading, not compiled, not run on hardware
//
#include <hip/hip_runtime.h>

// MHA for B=4, L=2048, D=1024, H=16, HD=64 (fp32 in/out, bf16 WMMA internally).

#define BB 4
#define LL 2048
#define DD 1024
#define HH 16
#define HD 64

typedef __bf16 bf16;
typedef __attribute__((ext_vector_type(16))) __bf16 v16bf;
typedef __attribute__((ext_vector_type(8)))  __bf16 v8bf;
typedef __attribute__((ext_vector_type(8)))  float  v8f;

// ---- CDNA5 async global->LDS copy (ASYNCcnt) with sync fallback ----
#if defined(__has_builtin)
#if __has_builtin(__builtin_amdgcn_global_load_async_to_lds_b128) && \
    __has_builtin(__builtin_amdgcn_s_wait_asynccnt)
#define HAVE_ASYNC_LDS 1
#endif
#endif

#ifdef HAVE_ASYNC_LDS
typedef int v4i __attribute__((vector_size(16)));
#define AS1 __attribute__((address_space(1)))
#define AS3 __attribute__((address_space(3)))
static __device__ __forceinline__ void cp16(void* lds, const void* g) {
  // per-lane 16B: one wave-level global_load_async_to_lds_b128
  __builtin_amdgcn_global_load_async_to_lds_b128(
      (AS1 v4i*)(unsigned long long)g, (AS3 v4i*)lds, 0, 0);
}
static __device__ __forceinline__ void cp_wait() {
  __builtin_amdgcn_s_wait_asynccnt(0);
}
#else
static __device__ __forceinline__ void cp16(void* lds, const void* g) {
  *(v8bf*)lds = *(const v8bf*)g;
}
static __device__ __forceinline__ void cp_wait() {}
#endif

// ---- WMMA fragment helpers (CDNA5 ISA 7.12.2 layouts, wave32) ----

// A fragment: 16x32 (MxK) bf16 from a row-major tile, row stride ld (elements).
static __device__ __forceinline__ v16bf load_fragA(const bf16* base, int ld) {
  const int lane = threadIdx.x & 31;
  const int row  = lane & 15;
  const int kb   = (lane >> 4) << 3;
  const bf16* p  = base + row * ld + kb;
  v8bf lo = *(const v8bf*)p;
  v8bf hi = *(const v8bf*)(p + 16);
  v16bf r;
#pragma unroll
  for (int i = 0; i < 8; ++i) { r[i] = lo[i]; r[i + 8] = hi[i]; }
  return r;
}

// B fragment: 32x16 (KxN) bf16 from a tile stored transposed [N][K], stride ld.
static __device__ __forceinline__ v16bf load_fragB(const bf16* base, int ld) {
  const int lane = threadIdx.x & 31;
  const int n    = lane & 15;
  const int kb   = (lane >> 4) << 3;
  const bf16* p  = base + n * ld + kb;
  v8bf lo = *(const v8bf*)p;
  v8bf hi = *(const v8bf*)(p + 16);
  v16bf r;
#pragma unroll
  for (int i = 0; i < 8; ++i) { r[i] = lo[i]; r[i + 8] = hi[i]; }
  return r;
}

static __device__ __forceinline__ v8f wmma_bf16(v16bf a, v16bf b, v8f c) {
  return __builtin_amdgcn_wmma_f32_16x16x32_bf16(false, a, false, b, (short)0, c,
                                                 false, false);
}

// ---- conversion kernels ----

__global__ void cvt_bf16_kernel(const float* __restrict__ x, bf16* __restrict__ y,
                                int n) {
  int i = blockIdx.x * blockDim.x + threadIdx.x;
  if (i < n) y[i] = (bf16)x[i];
}

// Wt[n][k] = W[k][n], D=1024
__global__ void transpose_w_kernel(const float* __restrict__ W,
                                   bf16* __restrict__ Wt) {
  int i = blockIdx.x * blockDim.x + threadIdx.x;
  int r = i >> 10, c = i & 1023;
  Wt[((size_t)c << 10) + r] = (bf16)W[i];
}

// ---- WMMA GEMM: Y = A[M,K] @ Wt[N,K]^T + bias, scatter per mode ----
// Double-buffered LDS, BK=64, async staging, one barrier per k-step.
// mode 1: bf16 out at [b,h,l,hd]; mode 2: bf16 out [b,h,hd,l]; mode 3: fp32 [M,N]
__global__ __launch_bounds__(256) void wmma_gemm_kernel(
    const bf16* __restrict__ A, const bf16* __restrict__ Bt,
    const float* __restrict__ bias, void* __restrict__ out,
    int M, int N, int Kd, int mode) {
  __shared__ bf16 As[2][128][64];
  __shared__ bf16 Bs[2][128][64];
  const int t    = threadIdx.x;
  const int lane = t & 31;
  const int wave = t >> 5;
  const int wm   = wave & 1;   // 2 waves over M (64 rows each)
  const int wn   = wave >> 1;  // 4 waves over N (32 cols each)
  const int m0   = blockIdx.y * 128;
  const int n0   = blockIdx.x * 128;

  v8f acc[4][2];
#pragma unroll
  for (int i = 0; i < 4; ++i)
#pragma unroll
    for (int j = 0; j < 2; ++j)
#pragma unroll
      for (int r = 0; r < 8; ++r) acc[i][j][r] = 0.f;

  // stage(buf, k0): 128x64 bf16 tiles of A and Bt; 4 chunks each per thread
  auto stage = [&](int buf, int k0) {
#pragma unroll
    for (int c = 0; c < 4; ++c) {
      int idx = t + c * 256;
      int row = idx >> 3;
      int col = (idx & 7) << 3;
      cp16(&As[buf][row][col], &A[(size_t)(m0 + row) * Kd + k0 + col]);
      cp16(&Bs[buf][row][col], &Bt[(size_t)(n0 + row) * Kd + k0 + col]);
    }
  };

  const int nIter = Kd >> 6;  // BK = 64
  stage(0, 0);
  for (int it = 0; it < nIter; ++it) {
    const int cur = it & 1;
    cp_wait();
    __syncthreads();  // all waves' async stage of `cur` complete & visible
    if (it + 1 < nIter) stage(1 - cur, (it + 1) << 6);
#pragma unroll
    for (int ks = 0; ks < 64; ks += 32) {
      v16bf bfr[2];
#pragma unroll
      for (int j = 0; j < 2; ++j)
        bfr[j] = load_fragB(&Bs[cur][wn * 32 + j * 16][ks], 64);
#pragma unroll
      for (int i = 0; i < 4; ++i) {
        v16bf af = load_fragA(&As[cur][wm * 64 + i * 16][ks], 64);
#pragma unroll
        for (int j = 0; j < 2; ++j) acc[i][j] = wmma_bf16(af, bfr[j], acc[i][j]);
      }
    }
  }

  const int nLane = lane & 15;
  const int mHalf = (lane >> 4) << 3;
#pragma unroll
  for (int j = 0; j < 2; ++j) {
    const int n  = n0 + wn * 32 + j * 16 + nLane;
    const float bz = bias[n];
#pragma unroll
    for (int i = 0; i < 4; ++i) {
#pragma unroll
      for (int r = 0; r < 8; ++r) {
        const int m = m0 + wm * 64 + i * 16 + mHalf + r;
        float v = acc[i][j][r] + bz;
        if (mode == 1) {
          int b = m >> 11, l = m & 2047, h = n >> 6, hd = n & 63;
          ((bf16*)out)[((size_t)((b * 16 + h) * 2048 + l) << 6) + hd] = (bf16)v;
        } else if (mode == 2) {
          int b = m >> 11, l = m & 2047, h = n >> 6, hd = n & 63;
          ((bf16*)out)[((size_t)((b * 16 + h) * 64 + hd) << 11) + l] = (bf16)v;
        } else {
          ((float*)out)[(size_t)m * N + n] = v;
        }
      }
    }
  }
}

// ---- Flash attention ----
// Q,K: [B,H,L,HD] bf16; Vt: [B,H,HD,L] bf16; O: [B,L,D] bf16.
// Block = 128 queries of one (b,h); 8 waves x 16 queries. 64-key blocks,
// double-buffered K/V tiles, one barrier per block.
// S^T = K_blk @ Q^T so the softmax reduction is a lane^16 shuffle.
__global__ __launch_bounds__(256) void fa_kernel(
    const bf16* __restrict__ Q, const bf16* __restrict__ K,
    const bf16* __restrict__ Vt, bf16* __restrict__ O) {
  __shared__ bf16 Ks[2][64][64];    // [key][hd]
  __shared__ bf16 Vs[2][64][64];    // [hd][key]
  __shared__ bf16 Ps[8][16][64];    // per-wave P: [query][key]

  const int t    = threadIdx.x;
  const int lane = t & 31;
  const int wave = t >> 5;
  const int bh   = blockIdx.y;      // b*16 + h
  const int b    = bh >> 4, h = bh & 15;
  const int q0   = blockIdx.x * 128 + wave * 16;

  const bf16* Qh = Q + ((size_t)bh << 11) * 64;    // [L][64]
  const bf16* Kh = K + ((size_t)bh << 11) * 64;    // [L][64]
  const bf16* Vh = Vt + ((size_t)bh << 6) * 2048;  // [64][L]

  // Q strip as B fragments ([n=query][k=hd] row-major = Qh rows)
  v16bf qf0 = load_fragB(Qh + (size_t)q0 * 64, 64);
  v16bf qf1 = load_fragB(Qh + (size_t)q0 * 64 + 32, 64);

  float m_run = -__builtin_inff();
  float l_run = 0.f;
  v8f oacc[4];
#pragma unroll
  for (int i = 0; i < 4; ++i)
#pragma unroll
    for (int r = 0; r < 8; ++r) oacc[i][r] = 0.f;

  const float scale = 0.125f;  // 1/sqrt(64)

  auto stage = [&](int buf, int j0) {
#pragma unroll
    for (int c = 0; c < 2; ++c) {
      int idx = t + c * 256;
      int row = idx >> 3;
      int col = (idx & 7) << 3;
      cp16(&Ks[buf][row][col], &Kh[(size_t)(j0 + row) * 64 + col]);
      cp16(&Vs[buf][row][col], &Vh[((size_t)row << 11) + j0 + col]);
    }
  };

  stage(0, 0);
  for (int j0 = 0; j0 < LL; j0 += 64) {
    const int cur = (j0 >> 6) & 1;
    cp_wait();
    __syncthreads();
    if (j0 + 64 < LL) stage(1 - cur, j0 + 64);

    // S^T (64 keys x 16 queries) = K_blk(64x64) @ Q^T(64x16)
    v8f s[4];
#pragma unroll
    for (int i = 0; i < 4; ++i) {
      v8f c;
#pragma unroll
      for (int r = 0; r < 8; ++r) c[r] = 0.f;
      v16bf a0 = load_fragA(&Ks[cur][i * 16][0], 64);
      v16bf a1 = load_fragA(&Ks[cur][i * 16][32], 64);
      c = wmma_bf16(a0, qf0, c);
      c = wmma_bf16(a1, qf1, c);
      s[i] = c;
    }

    // online softmax: lane and lane^16 hold the same query's key-halves
    float mloc = -__builtin_inff();
#pragma unroll
    for (int i = 0; i < 4; ++i)
#pragma unroll
      for (int r = 0; r < 8; ++r) mloc = fmaxf(mloc, s[i][r]);
    mloc *= scale;
    mloc = fmaxf(mloc, __shfl_xor(mloc, 16));
    float m_new = fmaxf(m_run, mloc);
    float corr  = __expf(m_run - m_new);
    l_run *= corr;
#pragma unroll
    for (int i = 0; i < 4; ++i)
#pragma unroll
      for (int r = 0; r < 8; ++r) oacc[i][r] *= corr;

    const int qn    = lane & 15;
    const int kHalf = (lane >> 4) << 3;
    float lsum = 0.f;
#pragma unroll
    for (int i = 0; i < 4; ++i) {
#pragma unroll
      for (int r = 0; r < 8; ++r) {
        float p = __expf(s[i][r] * scale - m_new);
        lsum += p;
        Ps[wave][qn][i * 16 + kHalf + r] = (bf16)p;  // transposed store: [q][key]
      }
    }
    l_run += lsum;
    m_run = m_new;

    // O^T (64 hd x 16 q) += V^T_blk(64x64) @ P^T(64x16); DS in-order per wave
    v16bf bp0 = load_fragB(&Ps[wave][0][0], 64);
    v16bf bp1 = load_fragB(&Ps[wave][0][32], 64);
#pragma unroll
    for (int i = 0; i < 4; ++i) {
      v16bf a0 = load_fragA(&Vs[cur][i * 16][0], 64);
      v16bf a1 = load_fragA(&Vs[cur][i * 16][32], 64);
      oacc[i] = wmma_bf16(a0, bp0, oacc[i]);
      oacc[i] = wmma_bf16(a1, bp1, oacc[i]);
    }
  }

  float l_tot = l_run + __shfl_xor(l_run, 16);
  float inv = 1.f / l_tot;
  const int qn    = lane & 15;
  const int hHalf = (lane >> 4) << 3;
#pragma unroll
  for (int i = 0; i < 4; ++i)
#pragma unroll
    for (int r = 0; r < 8; ++r) {
      int hd = i * 16 + hHalf + r;
      O[((size_t)(b * 2048 + q0 + qn) << 10) + h * 64 + hd] =
          (bf16)(oacc[i][r] * inv);
    }
}

// ---- host launch ----

extern "C" void kernel_launch(void* const* d_in, const int* in_sizes, int n_in,
                              void* d_out, int out_size, void* d_ws, size_t ws_size,
                              hipStream_t stream) {
  const float* q  = (const float*)d_in[0];
  const float* Wq = (const float*)d_in[1];
  const float* bq = (const float*)d_in[2];
  const float* Wk = (const float*)d_in[3];
  const float* bk = (const float*)d_in[4];
  const float* Wv = (const float*)d_in[5];
  const float* bv = (const float*)d_in[6];
  const float* Wo = (const float*)d_in[7];
  const float* bo = (const float*)d_in[8];

  char* ws = (char*)d_ws;
  size_t off = 0;
  auto alloc = [&](size_t bytes) -> void* {
    void* p = ws + off;
    off = (off + bytes + 255) & ~(size_t)255;
    return p;
  };
  const size_t nX = (size_t)BB * LL * DD;  // 8,388,608
  bf16* Xb  = (bf16*)alloc(nX * 2);
  bf16* Wqt = (bf16*)alloc((size_t)DD * DD * 2);
  bf16* Wkt = (bf16*)alloc((size_t)DD * DD * 2);
  bf16* Wvt = (bf16*)alloc((size_t)DD * DD * 2);
  bf16* Wot = (bf16*)alloc((size_t)DD * DD * 2);
  bf16* Qb  = (bf16*)alloc(nX * 2);
  bf16* Kb  = (bf16*)alloc(nX * 2);
  bf16* Vtb = (bf16*)alloc(nX * 2);
  bf16* Ab  = (bf16*)alloc(nX * 2);

  cvt_bf16_kernel<<<(int)(nX / 256), 256, 0, stream>>>(q, Xb, (int)nX);
  transpose_w_kernel<<<(DD * DD) / 256, 256, 0, stream>>>(Wq, Wqt);
  transpose_w_kernel<<<(DD * DD) / 256, 256, 0, stream>>>(Wk, Wkt);
  transpose_w_kernel<<<(DD * DD) / 256, 256, 0, stream>>>(Wv, Wvt);
  transpose_w_kernel<<<(DD * DD) / 256, 256, 0, stream>>>(Wo, Wot);

  dim3 gg(DD / 128, (BB * LL) / 128);  // (8, 64)
  wmma_gemm_kernel<<<gg, 256, 0, stream>>>(Xb, Wqt, bq, Qb, BB * LL, DD, DD, 1);
  wmma_gemm_kernel<<<gg, 256, 0, stream>>>(Xb, Wkt, bk, Kb, BB * LL, DD, DD, 1);
  wmma_gemm_kernel<<<gg, 256, 0, stream>>>(Xb, Wvt, bv, Vtb, BB * LL, DD, DD, 2);

  fa_kernel<<<dim3(LL / 128, BB * HH), 256, 0, stream>>>(Qb, Kb, Vtb, Ab);

  wmma_gemm_kernel<<<gg, 256, 0, stream>>>(Ab, Wot, bo, d_out, BB * LL, DD, DD, 3);
}
